// PVConv_57217554317615
// MI455X (gfx1250) — compile-verified, matrix operand order checked by reading, not run on hardware
//
#include <hip/hip_runtime.h>

// ---------------- problem constants (from reference setup_inputs) ----------
#define BATCH 8
#define CIN1  64
#define COUT  128
#define RES   32
#define NVOX  32768       // 32^3
#define NPTS  32768
#define GN_EPS 1e-5f

typedef __attribute__((ext_vector_type(16))) __bf16       bf16x16;
typedef __attribute__((ext_vector_type(8)))  float        f32x8;
typedef __attribute__((ext_vector_type(4)))  unsigned int u32x4;

union BFrag { bf16x16 v; u32x4 u[2]; };

__device__ __forceinline__ __bf16 tobf(float f) {
  union { float f; unsigned u; } a; a.f = f;
  unsigned r = (a.u + 0x7FFFu + ((a.u >> 16) & 1u)) >> 16;   // RNE
  union { unsigned short s; __bf16 h; } b; b.s = (unsigned short)r;
  return b.h;
}
__device__ __forceinline__ int imin(int a, int b) { return a < b ? a : b; }

// ---------------- coords: mean, scale, normalized coords -------------------
__device__ __forceinline__ float blockSum(float v, float* red) {
  red[threadIdx.x] = v; __syncthreads();
  for (int s = 128; s > 0; s >>= 1) {
    if ((int)threadIdx.x < s) red[threadIdx.x] += red[threadIdx.x + s];
    __syncthreads();
  }
  float r = red[0]; __syncthreads(); return r;
}
__device__ __forceinline__ float blockMax(float v, float* red) {
  red[threadIdx.x] = v; __syncthreads();
  for (int s = 128; s > 0; s >>= 1) {
    if ((int)threadIdx.x < s) red[threadIdx.x] = fmaxf(red[threadIdx.x], red[threadIdx.x + s]);
    __syncthreads();
  }
  float r = red[0]; __syncthreads(); return r;
}

__global__ __launch_bounds__(256)
void coords_stats_kernel(const float* __restrict__ C, float* __restrict__ NC) {
  __shared__ float red[256];
  const int b = blockIdx.x;
  const float* cx = C + (size_t)b * 3 * NPTS;
  const float* cy = cx + NPTS;
  const float* cz = cy + NPTS;
  float sx = 0.f, sy = 0.f, sz = 0.f;
  for (int n = threadIdx.x; n < NPTS; n += 256) { sx += cx[n]; sy += cy[n]; sz += cz[n]; }
  const float mx = blockSum(sx, red) * (1.f / NPTS);
  const float my = blockSum(sy, red) * (1.f / NPTS);
  const float mz = blockSum(sz, red) * (1.f / NPTS);
  float mv = 0.f;
  for (int n = threadIdx.x; n < NPTS; n += 256) {
    float dx = cx[n] - mx, dy = cy[n] - my, dz = cz[n] - mz;
    mv = fmaxf(mv, sqrtf(dx * dx + dy * dy + dz * dz));
  }
  const float inv = 1.f / (blockMax(mv, red) * 2.f);
  float* ox = NC + (size_t)b * 3 * NPTS;
  float* oy = ox + NPTS;
  float* oz = oy + NPTS;
  for (int n = threadIdx.x; n < NPTS; n += 256) {
    ox[n] = fminf(fmaxf(((cx[n] - mx) * inv + 0.5f) * (float)RES, 0.f), (float)(RES - 1));
    oy[n] = fminf(fmaxf(((cy[n] - my) * inv + 0.5f) * (float)RES, 0.f), (float)(RES - 1));
    oz[n] = fminf(fmaxf(((cz[n] - mz) * inv + 0.5f) * (float)RES, 0.f), (float)(RES - 1));
  }
}

// ---------------- scatter-mean voxelization --------------------------------
__global__ __launch_bounds__(256)
void zero_f32(float* __restrict__ p, size_t n) {
  size_t i = (size_t)blockIdx.x * 256 + threadIdx.x;
  if (i < n) p[i] = 0.f;
}

__global__ __launch_bounds__(256)
void scatter_kernel(const float* __restrict__ F, const float* __restrict__ NC,
                    float* __restrict__ GSUM, float* __restrict__ CNT) {
  const int t = blockIdx.x * 256 + threadIdx.x;
  const int n = t & (NPTS - 1);
  const int b = t >> 15;
  const int vx = (int)rintf(NC[((size_t)b * 3 + 0) * NPTS + n]);
  const int vy = (int)rintf(NC[((size_t)b * 3 + 1) * NPTS + n]);
  const int vz = (int)rintf(NC[((size_t)b * 3 + 2) * NPTS + n]);
  const int flat = (vx * RES + vy) * RES + vz;
  float* gp = GSUM + ((size_t)b * NVOX + flat) * CIN1;
  unsafeAtomicAdd(CNT + (size_t)b * NVOX + flat, 1.f);   // global_atomic_add_f32
  #pragma unroll 4
  for (int c = 0; c < CIN1; ++c)
    unsafeAtomicAdd(gp + c, F[((size_t)b * CIN1 + c) * NPTS + n]);
}

// sums -> mean, f32 -> bf16, layout [b][voxel][cin] (channel-fastest for WMMA B)
__global__ __launch_bounds__(256)
void finalize_grid_kernel(const float* __restrict__ GSUM, const float* __restrict__ CNT,
                          __bf16* __restrict__ G0) {
  const size_t id = (size_t)blockIdx.x * 256 + threadIdx.x;   // (b*NVOX+v)*64 + c
  const size_t bv = id >> 6;
  const float d = fmaxf(CNT[bv], 1.f);
  G0[id] = tobf(GSUM[id] / d);
}

// ---------------- weight pre-swizzle into WMMA A-fragment layout -----------
// A 16x32 bf16 (ISA 7.12.2): lane<16 -> M=lane, K in {0..7,16..23};
// lane>=16 -> M=lane-16, K in {8..15,24..31}. elem e: K = koff + (e&7) + (e&8 ? 16:0)
__global__ __launch_bounds__(256)
void swizzle_w_kernel(const float* __restrict__ w, __bf16* __restrict__ out,
                      int cin, int taps, int total) {
  const int id = blockIdx.x * 256 + threadIdx.x;
  if (id >= total) return;
  const int lane = id & 31;
  const int fr = id >> 5;                 // frag = (tap*chunks + ch)*8 + oc_tile
  const int oct = fr & 7;
  const int rest = fr >> 3;
  const int chunks = cin >> 5;
  const int ch = rest % chunks;
  const int tap = rest / chunks;
  const int oc = oct * 16 + (lane & 15);
  const int koff = (lane >> 4) << 3;
  __bf16* op = out + (size_t)fr * 512 + lane * 16;
  #pragma unroll
  for (int e = 0; e < 16; ++e) {
    const int k = koff + (e & 7) + ((e & 8) ? 16 : 0);
    const int ci = ch * 32 + k;
    op[e] = tobf(w[((size_t)oc * cin + ci) * taps + tap]);
  }
}

// ---------------- implicit-GEMM conv3d / GEMM via WMMA bf16 ----------------
// Block: 256 thr = 8 wave32. Tile: 32 voxels (one z-row) x 128 out-ch.
// Wave w: voxel sub-tile (w>>2)*16..+15, oc tiles {2*(w&3), 2*(w&3)+1}.
// Two independent accumulators share each B fragment -> 2 back-to-back
// independent v_wmma_f32_16x16x32_bf16 per K-chunk (fills WMMA->WMMA RAW
// hazard slots, halves B traffic per WMMA). Out-of-bounds columns are
// redirected to a zero pad via address cndmask (no EXEC branching in the
// inner loop). EXEC is all-1s at every WMMA; the x/y tap skip is uniform.
template <int CIN, bool CONV3>
__global__ __launch_bounds__(256)
void wmma_conv(const __bf16* __restrict__ X, const __bf16* __restrict__ WA,
               const __bf16* __restrict__ ZPAD, const float* __restrict__ bias,
               float* __restrict__ Y) {
  constexpr int CH = CIN / 32;
  constexpr int TAPS = CONV3 ? 27 : 1;
  const int lane = threadIdx.x & 31;
  const int wv = threadIdx.x >> 5;
  const int sub = wv >> 2;                  // voxel sub-tile 0/1
  const int op2 = (wv & 3) * 2;             // first of two oc tiles
  const int t = blockIdx.x & 1023;          // 1024 z-rows (or point tiles) per batch
  const int b = blockIdx.x >> 10;
  int x0 = 0, y0 = 0;
  if constexpr (CONV3) { x0 = t >> 5; y0 = t & 31; }
  const int ncol = lane & 15;               // column within 16-wide sub-tile
  const int zcol = sub * 16 + ncol;         // z (or point offset) within 32-row
  const int koff = (lane >> 4) << 3;        // B-frag K offset: 0 or 8
  f32x8 acc0 = {}, acc1 = {};

  for (int tap = 0; tap < TAPS; ++tap) {
    const __bf16* xp;
    if constexpr (CONV3) {
      const int dx = tap / 9 - 1, dy = (tap / 3) % 3 - 1, dz = tap % 3 - 1;
      const int xx = x0 + dx, yy = y0 + dy;
      if (xx < 0 || xx > 31 || yy < 0 || yy > 31) continue;  // uniform across wave
      const int zz = zcol + dz;
      const bool colOK = (zz >= 0 && zz < 32);
      const __bf16* xbase =
          X + ((size_t)b * NVOX + ((size_t)(xx * 32 + yy) * 32 + (colOK ? zz : 0))) * CIN;
      xp = colOK ? xbase : ZPAD;            // v_cndmask on address, loads stay unconditional
    } else {
      xp = X + ((size_t)b * NPTS + (t * 32 + zcol)) * CIN;
    }
    #pragma unroll
    for (int ch = 0; ch < CH; ++ch) {
      const size_t fbase = ((size_t)(tap * CH + ch) * 8 + op2) * 512 + lane * 16;
      const bf16x16 a0 = *reinterpret_cast<const bf16x16*>(WA + fbase);
      const bf16x16 a1 = *reinterpret_cast<const bf16x16*>(WA + fbase + 512);
      BFrag bu;   // elems 0..7 = cin koff..koff+7 ; elems 8..15 = cin koff+16..+23
      bu.u[0] = *reinterpret_cast<const u32x4*>(xp + ch * 32 + koff);
      bu.u[1] = *reinterpret_cast<const u32x4*>(xp + ch * 32 + koff + 16);
      acc0 = __builtin_amdgcn_wmma_f32_16x16x32_bf16(
          false, a0, false, bu.v, (short)0, acc0, false, false);
      acc1 = __builtin_amdgcn_wmma_f32_16x16x32_bf16(
          false, a1, false, bu.v, (short)0, acc1, false, false);
    }
  }

  // C/D layout: VGPR r, lanes<16 -> (M=r, N=lane); lanes>=16 -> (M=r+8, N=lane-16)
  const int vout = t * 32 + zcol;
  const int hi8 = (lane >> 4) * 8;
  float* yrow = Y + ((size_t)b * NVOX + vout) * COUT;
  #pragma unroll
  for (int q = 0; q < 2; ++q) {
    const f32x8 acc = q ? acc1 : acc0;
    const int ocb = (op2 + q) * 16 + hi8;   // 8 consecutive oc
    float* yp = yrow + ocb;
    const float4 o0 = make_float4(acc[0] + bias[ocb + 0], acc[1] + bias[ocb + 1],
                                  acc[2] + bias[ocb + 2], acc[3] + bias[ocb + 3]);
    const float4 o1 = make_float4(acc[4] + bias[ocb + 4], acc[5] + bias[ocb + 5],
                                  acc[6] + bias[ocb + 6], acc[7] + bias[ocb + 7]);
    *reinterpret_cast<float4*>(yp) = o0;
    *reinterpret_cast<float4*>(yp + 4) = o1;
  }
}

// ---------------- GroupNorm statistics (8 groups of 16 ch over 32768) ------
__global__ __launch_bounds__(256)
void gn_stats_kernel(const float* __restrict__ X, float* __restrict__ mean,
                     float* __restrict__ rstd, float* __restrict__ chanmean) {
  __shared__ float sh[18];
  const int g = blockIdx.x & 7, b = blockIdx.x >> 3;
  if (threadIdx.x < 18) sh[threadIdx.x] = 0.f;
  __syncthreads();
  const float* xp = X + (size_t)b * NVOX * COUT + g * 16;
  float s = 0.f, s2 = 0.f, cs[16];
  #pragma unroll
  for (int q = 0; q < 16; ++q) cs[q] = 0.f;
  for (int v = threadIdx.x; v < NVOX; v += 256) {
    const float4* p = reinterpret_cast<const float4*>(xp + (size_t)v * COUT);
    #pragma unroll
    for (int q = 0; q < 4; ++q) {
      const float4 f = p[q];
      s  += f.x + f.y + f.z + f.w;
      s2 += f.x * f.x + f.y * f.y + f.z * f.z + f.w * f.w;
      cs[q * 4 + 0] += f.x; cs[q * 4 + 1] += f.y;
      cs[q * 4 + 2] += f.z; cs[q * 4 + 3] += f.w;
    }
  }
  atomicAdd(&sh[0], s);        // ds_add_f32
  atomicAdd(&sh[1], s2);
  if (chanmean) {
    #pragma unroll
    for (int q = 0; q < 16; ++q) atomicAdd(&sh[2 + q], cs[q]);
  }
  __syncthreads();
  if (threadIdx.x == 0) {
    const float inv = 1.f / (16.f * NVOX);
    const float m = sh[0] * inv;
    const float var = sh[1] * inv - m * m;
    mean[b * 8 + g] = m;
    rstd[b * 8 + g] = rsqrtf(var + GN_EPS);
  }
  if (chanmean && threadIdx.x < 16)
    chanmean[b * COUT + g * 16 + threadIdx.x] = sh[2 + threadIdx.x] * (1.f / NVOX);
}

// ---------------- GN + swish -> bf16 (conv1 epilogue) ----------------------
__global__ __launch_bounds__(256)
void gn_swish_bf16_kernel(const float* __restrict__ X, __bf16* __restrict__ Y,
                          const float* __restrict__ mean, const float* __restrict__ rstd,
                          const float* __restrict__ gamma, const float* __restrict__ beta) {
  const size_t i = ((size_t)blockIdx.x * 256 + threadIdx.x) * 4;
  const int b = (int)(i >> 22);               // NVOX*COUT = 2^22
  const int c0 = (int)(i & 127);
  const int g = c0 >> 4;
  const float m = mean[b * 8 + g], rs = rstd[b * 8 + g];
  const float4 x = *reinterpret_cast<const float4*>(X + i);
  const float xs[4] = {x.x, x.y, x.z, x.w};
  union { __bf16 h[4]; unsigned long long q; } o;
  #pragma unroll
  for (int j = 0; j < 4; ++j) {
    const float y = (xs[j] - m) * rs * gamma[c0 + j] + beta[c0 + j];
    o.h[j] = tobf(y / (1.f + __expf(-y)));
  }
  *reinterpret_cast<unsigned long long*>(Y + i) = o.q;
}

// ---------------- SE: tiny 2-layer MLP on post-GN channel means ------------
__global__ __launch_bounds__(128)
void se_kernel(const float* __restrict__ chanmean, const float* __restrict__ mean2,
               const float* __restrict__ rstd2, const float* __restrict__ g2g,
               const float* __restrict__ g2b, const float* __restrict__ w1,
               const float* __restrict__ b1, const float* __restrict__ w2,
               const float* __restrict__ b2, float* __restrict__ sescale) {
  __shared__ float m[COUT], s1[16];
  const int b = blockIdx.x, c = threadIdx.x;
  const int g = c >> 4;
  m[c] = (chanmean[b * COUT + c] - mean2[b * 8 + g]) * rstd2[b * 8 + g] * g2g[c] + g2b[c];
  __syncthreads();
  if (c < 16) {
    float a = b1[c];
    for (int j = 0; j < COUT; ++j) a += w1[c * COUT + j] * m[j];
    s1[c] = fmaxf(a, 0.f);
  }
  __syncthreads();
  float a = b2[c];
  #pragma unroll
  for (int j = 0; j < 16; ++j) a += w2[c * 16 + j] * s1[j];
  sescale[b * COUT + c] = 1.f / (1.f + __expf(-a));
}

// ---------------- GN2 + SE scale applied in place --------------------------
__global__ __launch_bounds__(256)
void gn_se_kernel(float* __restrict__ X, const float* __restrict__ mean,
                  const float* __restrict__ rstd, const float* __restrict__ gamma,
                  const float* __restrict__ beta, const float* __restrict__ sescale) {
  const size_t i = ((size_t)blockIdx.x * 256 + threadIdx.x) * 4;
  const int b = (int)(i >> 22);
  const int c0 = (int)(i & 127);
  const int g = c0 >> 4;
  const float m = mean[b * 8 + g], rs = rstd[b * 8 + g];
  const float4 x = *reinterpret_cast<const float4*>(X + i);
  const float xs[4] = {x.x, x.y, x.z, x.w};
  float r[4];
  #pragma unroll
  for (int j = 0; j < 4; ++j)
    r[j] = ((xs[j] - m) * rs * gamma[c0 + j] + beta[c0 + j]) * sescale[b * COUT + c0 + j];
  *reinterpret_cast<float4*>(X + i) = make_float4(r[0], r[1], r[2], r[3]);
}

// ---------------- features [b][c][n] f32 -> [b][n][c] bf16 -----------------
__global__ __launch_bounds__(256)
void transpose_feats_kernel(const float* __restrict__ F, __bf16* __restrict__ FT) {
  const size_t id = (size_t)blockIdx.x * 256 + threadIdx.x;  // (b*64+c)*N + n
  const int n = (int)(id & (NPTS - 1));
  const int bc = (int)(id >> 15);
  const int c = bc & 63;
  const int b = bc >> 6;
  FT[((size_t)b * NPTS + n) * CIN1 + c] = tobf(F[id]);
}

// ---------------- trilinear devoxelize + point-branch GN+swish + add -------
__global__ __launch_bounds__(128)
void final_kernel(const float* __restrict__ NC, const float* __restrict__ G2,
                  const float* __restrict__ PT, const float* __restrict__ meanp,
                  const float* __restrict__ rstdp, const float* __restrict__ pg,
                  const float* __restrict__ pb, float* __restrict__ out) {
  const int b = blockIdx.x >> 8;
  const int n = ((blockIdx.x & 255) << 7) + threadIdx.x;
  float c3[3]; int lo[3], hi[3]; float fr[3];
  #pragma unroll
  for (int d = 0; d < 3; ++d) {
    c3[d] = NC[((size_t)b * 3 + d) * NPTS + n];
    lo[d] = (int)floorf(c3[d]);
    hi[d] = imin(lo[d] + 1, RES - 1);
    fr[d] = c3[d] - (float)lo[d];
  }
  int vidx[8]; float w8[8];
  #pragma unroll
  for (int k = 0; k < 8; ++k) {
    const int dx = (k >> 2) & 1, dy = (k >> 1) & 1, dz = k & 1;
    const int X = dx ? hi[0] : lo[0];
    const int Y = dy ? hi[1] : lo[1];
    const int Z = dz ? hi[2] : lo[2];
    vidx[k] = (X * RES + Y) * RES + Z;
    w8[k] = (dx ? fr[0] : 1.f - fr[0]) * (dy ? fr[1] : 1.f - fr[1]) * (dz ? fr[2] : 1.f - fr[2]);
  }
  const float* gb = G2 + (size_t)b * NVOX * COUT;
  const float* pp = PT + ((size_t)b * NPTS + n) * COUT;
  for (int c = 0; c < COUT; ++c) {
    float acc = 0.f;
    #pragma unroll
    for (int k = 0; k < 8; ++k) acc = fmaf(w8[k], gb[(size_t)vidx[k] * COUT + c], acc);
    const int g = c >> 4;
    float y = (pp[c] - meanp[b * 8 + g]) * rstdp[b * 8 + g] * pg[c] + pb[c];
    y = y / (1.f + __expf(-y));
    out[((size_t)b * COUT + c) * NPTS + n] = acc + y;
  }
}

// ---------------------------------------------------------------------------
extern "C" void kernel_launch(void* const* d_in, const int* in_sizes, int n_in,
                              void* d_out, int out_size, void* d_ws, size_t ws_size,
                              hipStream_t stream) {
  (void)in_sizes; (void)n_in; (void)out_size; (void)ws_size;
  const float* features = (const float*)d_in[0];
  const float* coords   = (const float*)d_in[1];
  const float* conv1_w  = (const float*)d_in[2];
  const float* conv1_b  = (const float*)d_in[3];
  const float* gn1_g    = (const float*)d_in[4];
  const float* gn1_b    = (const float*)d_in[5];
  const float* conv2_w  = (const float*)d_in[6];
  const float* conv2_b  = (const float*)d_in[7];
  const float* gn2_g    = (const float*)d_in[8];
  const float* gn2_b    = (const float*)d_in[9];
  const float* se_w1    = (const float*)d_in[10];
  const float* se_b1    = (const float*)d_in[11];
  const float* se_w2    = (const float*)d_in[12];
  const float* se_b2    = (const float*)d_in[13];
  const float* pt_w     = (const float*)d_in[14];
  const float* pt_b     = (const float*)d_in[15];
  const float* ptgn_g   = (const float*)d_in[16];
  const float* ptgn_b   = (const float*)d_in[17];
  float* out = (float*)d_out;

  char* ws = (char*)d_ws;
  size_t off = 0;
  auto take = [&](size_t bytes) -> char* {
    char* p = ws + off;
    off = (off + bytes + 255) & ~(size_t)255;
    return p;
  };
  float*  NC    = (float*)take((size_t)BATCH * 3 * NPTS * 4);
  float*  CNT   = (float*)take((size_t)BATCH * NVOX * 4);
  float*  stats = (float*)take(16384);
  float *mean1 = stats,       *rstd1 = stats + 64,
        *mean2 = stats + 128, *rstd2 = stats + 192,
        *meanp = stats + 256, *rstdp = stats + 320,
        *chanmean = stats + 384, *sescale = stats + 1408;
  __bf16* ZPAD = (__bf16*)take(512);           // zero pad for OOB conv columns
  __bf16* WA1  = (__bf16*)take((size_t)27 * 2 * 8 * 512 * 2);
  __bf16* WA2  = (__bf16*)take((size_t)27 * 4 * 8 * 512 * 2);
  __bf16* WAP  = (__bf16*)take((size_t)1 * 2 * 8 * 512 * 2);
  float*  GSUM = (float*)take((size_t)BATCH * NVOX * CIN1 * 4);
  __bf16* G0   = (__bf16*)take((size_t)BATCH * NVOX * CIN1 * 2);
  float*  G1R  = (float*)take((size_t)BATCH * NVOX * COUT * 4);
  __bf16* G1   = (__bf16*)take((size_t)BATCH * NVOX * COUT * 2);
  float*  G2   = (float*)take((size_t)BATCH * NVOX * COUT * 4);
  __bf16* FT    = (__bf16*)GSUM;   // alias: GSUM dead after finalize_grid
  float*  PTRAW = G1R;             // alias: G1R dead after gn_swish_bf16

  // 1) voxelization
  coords_stats_kernel<<<BATCH, 256, 0, stream>>>(coords, NC);
  zero_f32<<<1, 256, 0, stream>>>((float*)ZPAD, 128);
  zero_f32<<<65536, 256, 0, stream>>>(GSUM, (size_t)BATCH * NVOX * CIN1);
  zero_f32<<<1024, 256, 0, stream>>>(CNT, (size_t)BATCH * NVOX);
  scatter_kernel<<<(BATCH * NPTS) / 256, 256, 0, stream>>>(features, NC, GSUM, CNT);
  finalize_grid_kernel<<<65536, 256, 0, stream>>>(GSUM, CNT, G0);

  // 2) weight pre-swizzle into WMMA A-fragment layout
  swizzle_w_kernel<<<54, 256, 0, stream>>>(conv1_w, WA1, CIN1, 27, 27 * 2 * 8 * 32);
  swizzle_w_kernel<<<108, 256, 0, stream>>>(conv2_w, WA2, COUT, 27, 27 * 4 * 8 * 32);
  swizzle_w_kernel<<<2, 256, 0, stream>>>(pt_w, WAP, CIN1, 1, 1 * 2 * 8 * 32);

  // 3) conv1 (WMMA implicit GEMM) -> GN stats -> GN+swish -> bf16
  wmma_conv<CIN1, true><<<BATCH * 1024, 256, 0, stream>>>(G0, WA1, ZPAD, conv1_b, G1R);
  gn_stats_kernel<<<64, 256, 0, stream>>>(G1R, mean1, rstd1, nullptr);
  gn_swish_bf16_kernel<<<32768, 256, 0, stream>>>(G1R, G1, mean1, rstd1, gn1_g, gn1_b);

  // 4) point branch: transpose -> WMMA GEMM -> GN stats
  transpose_feats_kernel<<<65536, 256, 0, stream>>>(features, FT);
  wmma_conv<CIN1, false><<<BATCH * 1024, 256, 0, stream>>>(FT, WAP, ZPAD, pt_b, PTRAW);
  gn_stats_kernel<<<64, 256, 0, stream>>>(PTRAW, meanp, rstdp, nullptr);

  // 5) conv2 -> GN stats (+channel means) -> SE -> apply GN2*SE
  wmma_conv<COUT, true><<<BATCH * 1024, 256, 0, stream>>>(G1, WA2, ZPAD, conv2_b, G2);
  gn_stats_kernel<<<64, 256, 0, stream>>>(G2, mean2, rstd2, chanmean);
  se_kernel<<<BATCH, 128, 0, stream>>>(chanmean, mean2, rstd2, gn2_g, gn2_b,
                                       se_w1, se_b1, se_w2, se_b2, sescale);
  gn_se_kernel<<<32768, 256, 0, stream>>>(G2, mean2, rstd2, gn2_g, gn2_b, sescale);

  // 6) devoxelize + point GN+swish + add
  final_kernel<<<BATCH * 256, 128, 0, stream>>>(NC, G2, PTRAW, meanp, rstdp,
                                                ptgn_g, ptgn_b, out);
}
